// IsoMaxLossFirstPart_24696061952724
// MI455X (gfx1250) — matrix operand order repaired and checked by reading, not compiled, not run on hardware
//
#include <hip/hip_runtime.h>
#include <math.h>

// IsoMax logits on gfx1250: logits[n,c] = -sqrt(max(||x||^2 + ||p||^2 - 2 x.p, 0))
// Fast path: pre-pass splits f32 -> packed bf16 (hi|lo) rows + exact f32 norms,
// then a WMMA GEMM (3-pass bf16 split: hi*hi + hi*lo + lo*hi, f32 accumulate)
// whose hot loop is just b128 loads + v_wmma_f32_16x16x32_bf16 (no per-tile
// conversion VALU). Fallback (small ws): fused convert-in-loop kernel.

typedef __attribute__((ext_vector_type(16))) __bf16 v16bf;
typedef __attribute__((ext_vector_type(8)))  __bf16 v8bf;
typedef __attribute__((ext_vector_type(8)))  float  v8f;
typedef __attribute__((ext_vector_type(4)))  float  v4f;

namespace {
constexpr int kN = 65536;
constexpr int kD = 256;
constexpr int kC = 1000;
constexpr int kColsPad = 1024;        // 16 groups of 64 columns (tail masked)
constexpr int kRowsPerBlock = 128;    // 8 waves x 16 rows
constexpr int kColsPerBlock = 64;     // 4 WMMA tiles of 16 cols per wave
constexpr int kKHalf = 128;           // LDS stage granularity in K
constexpr int kPackRow = 2 * kD;      // packed bf16 row: hi[256] | lo[256]
constexpr int kLdsRow = kKHalf + 16;  // 144 bf16 = 288B row -> 8-bank stride
}

union FragU {
  v16bf v;
  struct { v8bf lo, hi; } p;
  __device__ FragU() {}
};

// ---------------------------------------------------------------------------
// Pre-pass: wave per row. f32 row -> bf16 hi/lo packed row + exact f32 norm.
// ---------------------------------------------------------------------------
__global__ __launch_bounds__(256)
void split_bf16_rows_kernel(const float* __restrict__ src,
                            __bf16* __restrict__ dst,
                            float* __restrict__ norms,
                            int nrows)
{
  const int wave = (int)((blockIdx.x * blockDim.x + threadIdx.x) >> 5);
  const int lane = threadIdx.x & 31;
  if (wave >= nrows) return;

  const float* s = src + (size_t)wave * kD + lane * 8;
  v4f f0 = *(const v4f*)(s + 0);
  v4f f1 = *(const v4f*)(s + 4);
  float fv[8];
  #pragma unroll
  for (int i = 0; i < 4; ++i) { fv[i] = f0[i]; fv[4 + i] = f1[i]; }

  v8bf hi, lo;
  float ss = 0.f;
  #pragma unroll
  for (int i = 0; i < 8; ++i) {
    float  f = fv[i];
    __bf16 h = (__bf16)f;
    hi[i] = h;
    lo[i] = (__bf16)(f - (float)h);
    ss = fmaf(f, f, ss);
  }

  __bf16* d = dst + (size_t)wave * kPackRow + lane * 8;
  *(v8bf*)(d)       = hi;
  *(v8bf*)(d + kD)  = lo;

  #pragma unroll
  for (int off = 16; off > 0; off >>= 1) ss += __shfl_xor(ss, off, 32);
  if (lane == 0) norms[wave] = ss;
}

// ---------------------------------------------------------------------------
// Fast-path GEMM: packed bf16 inputs, LDS-staged B, pure load+WMMA hot loop.
// ---------------------------------------------------------------------------
__global__ __launch_bounds__(256)
void isomax_wmma_packed_kernel(const __bf16* __restrict__ Ap,
                               const __bf16* __restrict__ Bp,
                               const float*  __restrict__ x2,
                               const float*  __restrict__ p2,
                               float* __restrict__ out)
{
  __shared__ __bf16 ldsHi[kColsPerBlock][kLdsRow];  // 18 KB
  __shared__ __bf16 ldsLo[kColsPerBlock][kLdsRow];  // 18 KB

  const int tid   = threadIdx.x;
  const int lane  = tid & 31;
  const int wave  = tid >> 5;
  const int lrow  = lane & 15;
  const int lhalf = lane >> 4;

  const int m0 = blockIdx.x * kRowsPerBlock + wave * 16;
  const int n0 = blockIdx.y * kColsPerBlock;

  const __bf16* aRow = Ap + (size_t)(m0 + lrow) * kPackRow;

  v8f acc[4] = {};

  for (int h = 0; h < 2; ++h) {     // two K halves of 128
    __syncthreads();                // prior half fully consumed
    // stage 64 rows x 128 K of hi and lo (16B chunks; 16 chunks per row)
    #pragma unroll
    for (int i = 0; i < 4; ++i) {
      int idx = i * 256 + tid;      // 0..1023
      int r   = idx >> 4;
      int c16 = idx & 15;
      int g   = n0 + r; g = g < kC ? g : (kC - 1);
      const __bf16* srow = Bp + (size_t)g * kPackRow + h * kKHalf + c16 * 8;
      *(v8bf*)&ldsHi[r][c16 * 8] = *(const v8bf*)(srow);
      *(v8bf*)&ldsLo[r][c16 * 8] = *(const v8bf*)(srow + kD);
    }
    __syncthreads();

    #pragma unroll
    for (int k0 = 0; k0 < kKHalf; k0 += 32) {
      const int k = h * kKHalf + k0;

      // A fragment (16-bit A 16x32 layout): K = k+8*lhalf+{0..7} and +16
      const __bf16* pa = aRow + k + lhalf * 8;
      FragU aHi, aLo;
      aHi.p.lo = *(const v8bf*)(pa);
      aHi.p.hi = *(const v8bf*)(pa + 16);
      aLo.p.lo = *(const v8bf*)(pa + kD);
      aLo.p.hi = *(const v8bf*)(pa + kD + 16);

      #pragma unroll
      for (int t = 0; t < 4; ++t) {
        // B fragment (16-bit B 32x16 layout): K = k0+8*lhalf+{0..7} and +16
        const __bf16* ph = &ldsHi[t * 16 + lrow][k0 + lhalf * 8];
        const __bf16* pl = &ldsLo[t * 16 + lrow][k0 + lhalf * 8];
        FragU bHi, bLo;
        bHi.p.lo = *(const v8bf*)(ph);
        bHi.p.hi = *(const v8bf*)(ph + 16);
        bLo.p.lo = *(const v8bf*)(pl);
        bLo.p.hi = *(const v8bf*)(pl + 16);

        acc[t] = __builtin_amdgcn_wmma_f32_16x16x32_bf16(
            false, aHi.v, false, bHi.v, (short)0, acc[t], false, false);
        acc[t] = __builtin_amdgcn_wmma_f32_16x16x32_bf16(
            false, aHi.v, false, bLo.v, (short)0, acc[t], false, false);
        acc[t] = __builtin_amdgcn_wmma_f32_16x16x32_bf16(
            false, aLo.v, false, bHi.v, (short)0, acc[t], false, false);
      }
    }
  }

  // epilogue: sq = ||x||^2 + ||p||^2 - 2*xp ; logits = -sqrt(max(sq,0))
  float p2v[4];
  #pragma unroll
  for (int t = 0; t < 4; ++t) {
    int col = n0 + t * 16 + lrow;
    int cc  = col < kC ? col : (kC - 1);
    p2v[t] = p2[cc];
  }
  #pragma unroll
  for (int v = 0; v < 8; ++v) {
    int   row = m0 + v + 8 * lhalf;   // C layout: VGPR v -> row v (+8 hi half)
    float xr  = x2[row];
    #pragma unroll
    for (int t = 0; t < 4; ++t) {
      int col = n0 + t * 16 + lrow;
      if (col < kC) {
        float sq = xr + p2v[t] - 2.0f * acc[t][v];
        sq = sq > 0.f ? sq : 0.f;
        out[(size_t)row * kC + col] = -sqrtf(sq);
      }
    }
  }
}

// ---------------------------------------------------------------------------
// Fallback (small workspace): fused convert-in-loop kernel (round-1 version).
// ---------------------------------------------------------------------------
__global__ __launch_bounds__(256)
void isomax_fused_kernel(const float* __restrict__ feat,
                         const float* __restrict__ proto,
                         float* __restrict__ out)
{
  constexpr int kLdsStrideF = kKHalf + 4;
  __shared__ float ldsB[kColsPerBlock][kLdsStrideF];

  const int tid   = threadIdx.x;
  const int lane  = tid & 31;
  const int wave  = tid >> 5;
  const int lrow  = lane & 15;
  const int lhalf = lane >> 4;

  const int m0 = blockIdx.x * kRowsPerBlock + wave * 16;
  const int n0 = blockIdx.y * kColsPerBlock;

  const float* featRow = feat + (size_t)(m0 + lrow) * kD;

  v8f   acc[4] = {};
  float x2p    = 0.f;
  float p2p[4] = {0.f, 0.f, 0.f, 0.f};

  for (int h = 0; h < 2; ++h) {
    __syncthreads();
    #pragma unroll
    for (int i = 0; i < 8; ++i) {
      int idx4 = i * 256 + tid;
      int r    = idx4 >> 5;
      int c4   = idx4 & 31;
      int g    = n0 + r; g = g < kC ? g : (kC - 1);
      v4f b = *(const v4f*)(feat == proto ? proto : proto + (size_t)g * kD + h * kKHalf + c4 * 4);
      b = *(const v4f*)(proto + (size_t)g * kD + h * kKHalf + c4 * 4);
      *(v4f*)&ldsB[r][c4 * 4] = b;
    }
    __syncthreads();

    #pragma unroll
    for (int k0 = 0; k0 < kKHalf; k0 += 32) {
      const int k = h * kKHalf + k0;
      const float* pa = featRow + k + lhalf * 8;
      v4f a0 = *(const v4f*)(pa +  0);
      v4f a1 = *(const v4f*)(pa +  4);
      v4f a2 = *(const v4f*)(pa + 16);
      v4f a3 = *(const v4f*)(pa + 20);
      float af[16];
      #pragma unroll
      for (int i = 0; i < 4; ++i) {
        af[i] = a0[i]; af[4 + i] = a1[i]; af[8 + i] = a2[i]; af[12 + i] = a3[i];
      }
      v16bf ahi, alo;
      #pragma unroll
      for (int i = 0; i < 16; ++i) {
        float  f = af[i];
        __bf16 hv = (__bf16)f;
        ahi[i] = hv;
        alo[i] = (__bf16)(f - (float)hv);
        x2p = fmaf(f, f, x2p);
      }
      #pragma unroll
      for (int t = 0; t < 4; ++t) {
        const float* pb = &ldsB[t * 16 + lrow][k0 + lhalf * 8];
        v4f b0 = *(const v4f*)(pb + 0);
        v4f b1 = *(const v4f*)(pb + 4);
        v4f b2 = *(const v4f*)(pb + 16);
        v4f b3 = *(const v4f*)(pb + 20);
        float bfv[16];
        #pragma unroll
        for (int i = 0; i < 4; ++i) {
          bfv[i] = b0[i]; bfv[4 + i] = b1[i]; bfv[8 + i] = b2[i]; bfv[12 + i] = b3[i];
        }
        v16bf bhi, blo;
        #pragma unroll
        for (int i = 0; i < 16; ++i) {
          float  f = bfv[i];
          __bf16 hv = (__bf16)f;
          bhi[i] = hv;
          blo[i] = (__bf16)(f - (float)hv);
          p2p[t] = fmaf(f, f, p2p[t]);
        }
        acc[t] = __builtin_amdgcn_wmma_f32_16x16x32_bf16(
            false, ahi, false, bhi, (short)0, acc[t], false, false);
        acc[t] = __builtin_amdgcn_wmma_f32_16x16x32_bf16(
            false, ahi, false, blo, (short)0, acc[t], false, false);
        acc[t] = __builtin_amdgcn_wmma_f32_16x16x32_bf16(
            false, alo, false, bhi, (short)0, acc[t], false, false);
      }
    }
  }

  float x2f = x2p + __shfl_xor(x2p, 16, 32);
  #pragma unroll
  for (int t = 0; t < 4; ++t) p2p[t] += __shfl_xor(p2p[t], 16, 32);

  #pragma unroll
  for (int v = 0; v < 8; ++v) {
    float xr  = __shfl(x2f, v + 8 * lhalf, 32);
    int   row = m0 + v + 8 * lhalf;
    #pragma unroll
    for (int t = 0; t < 4; ++t) {
      int col = n0 + t * 16 + lrow;
      if (col < kC) {
        float sq = xr + p2p[t] - 2.0f * acc[t][v];
        sq = sq > 0.f ? sq : 0.f;
        out[(size_t)row * kC + col] = -sqrtf(sq);
      }
    }
  }
}

// ---------------------------------------------------------------------------
extern "C" void kernel_launch(void* const* d_in, const int* in_sizes, int n_in,
                              void* d_out, int out_size, void* d_ws, size_t ws_size,
                              hipStream_t stream) {
  (void)in_sizes; (void)n_in; (void)out_size;
  const float* feat  = (const float*)d_in[0];   // [65536, 256] f32
  const float* proto = (const float*)d_in[1];   // [1000, 256]  f32
  float* out = (float*)d_out;                   // [65536, 1000] f32

  const size_t szA  = (size_t)kN * kPackRow * sizeof(__bf16);  // 64 MB
  const size_t szB  = (size_t)kC * kPackRow * sizeof(__bf16);  // 1 MB
  const size_t szX2 = (size_t)kN * sizeof(float);              // 256 KB
  const size_t szP2 = 4096;
  const size_t offA = 0, offB = offA + szA, offX2 = offB + szB, offP2 = offX2 + szX2;
  const size_t need = offP2 + szP2;

  dim3 grid(kN / kRowsPerBlock, kColsPad / kColsPerBlock);  // 512 x 16

  if (ws_size >= need && d_ws != nullptr) {
    __bf16* Ap = (__bf16*)((char*)d_ws + offA);
    __bf16* Bp = (__bf16*)((char*)d_ws + offB);
    float*  x2 = (float*)((char*)d_ws + offX2);
    float*  p2 = (float*)((char*)d_ws + offP2);

    split_bf16_rows_kernel<<<kN / 8, 256, 0, stream>>>(feat, Ap, x2, kN);
    split_bf16_rows_kernel<<<(kC + 7) / 8, 256, 0, stream>>>(proto, Bp, p2, kC);
    isomax_wmma_packed_kernel<<<grid, dim3(256), 0, stream>>>(Ap, Bp, x2, p2, out);
  } else {
    isomax_fused_kernel<<<grid, dim3(256), 0, stream>>>(feat, proto, out);
  }
}
